// ShiftWindowMSA_15513421873332
// MI455X (gfx1250) — matrix operand order repaired and checked by reading, not compiled, not run on hardware
//
#include <hip/hip_runtime.h>
#include <hip/hip_bf16.h>

typedef __attribute__((ext_vector_type(16))) _Float16 v16h;
typedef __attribute__((ext_vector_type(8)))  _Float16 v8h;
typedef __attribute__((ext_vector_type(8)))  float    v8f;

#define WMMA_F16(a, b, c) \
  __builtin_amdgcn_wmma_f32_16x16x32_f16(false, (a), false, (b), (short)0, (c), false, false)

static __device__ __forceinline__ v16h combine16(v8h lo, v8h hi) {
  v16h r;
#pragma unroll
  for (int i = 0; i < 8; ++i) { r[i] = lo[i]; r[i + 8] = hi[i]; }
  return r;
}

// Region id for shifted-window mask: Hp=Wp=128, win=8, shift=4 -> cuts at 120,124
static __device__ __forceinline__ int reg1(int v) {
  return (v < 120) ? 0 : ((v < 124) ? 1 : 2);
}

// rel-pos coordinate: c[p] = 15*(p/8) + (p%8)
static __device__ __forceinline__ int rpc(int p) { return 15 * (p >> 3) + (p & 7); }

// window-order q row -> source token index (applies roll(-4,-4) + window partition)
static __device__ __forceinline__ size_t map_q_row(int row) {
  int w = row >> 6, qi = row & 63;
  int b = w >> 8, wh = (w >> 4) & 15, ww = w & 15;
  int y = wh * 8 + (qi >> 3), x = ww * 8 + (qi & 7);
  int oy = (y + 4) & 127, ox = (x + 4) & 127;
  return ((size_t)b << 14) + (oy << 7) + ox;
}

// window-order kv row -> source row in kv (B, nf=2, 16384, C)
static __device__ __forceinline__ size_t map_kv_row(int row) {
  int w = row >> 7, r = row & 127;
  int f = r >> 6, qi = r & 63;
  int b = w >> 8, wh = (w >> 4) & 15, ww = w & 15;
  int y = wh * 8 + (qi >> 3), x = ww * 8 + (qi & 7);
  int oy = (y + 4) & 127, ox = (x + 4) & 127;
  return ((size_t)(b * 2 + f) << 14) + (oy << 7) + ox;
}

// ---------------------------------------------------------------------------
// Projection GEMM (fp32 gathered A  x  fp32 W  -> f16 out), K = 512.
// MODE 0: q proj   (N=512,  out0 = Qh, (acc+bias)*scale)
// MODE 1: kv proj  (N=1024, cols <512 -> Kh, cols >=512 -> Vh)
// Block: 256 thr = 8 waves; tile 128x128; each wave 32x64 (2x4 wmma accums).
// ---------------------------------------------------------------------------
template <int MODE, int N>
__global__ __launch_bounds__(256) void gemm_proj(const float* __restrict__ A,
                                                 const float* __restrict__ W,
                                                 const float* __restrict__ bias,
                                                 _Float16* __restrict__ out0,
                                                 _Float16* __restrict__ out1) {
  __shared__ __align__(16) _Float16 As[128][40];  // [m][k]
  __shared__ __align__(16) _Float16 Bs[128][40];  // [n][k]

  const int nbn = N / 128;
  const int bm = blockIdx.x / nbn;
  const int bn = blockIdx.x % nbn;
  const int tid = threadIdx.x;
  const int lane = tid & 31;
  const int wv = tid >> 5;
  const int wm = wv >> 1, wn = wv & 1;
  const int lr = lane & 15, hi = lane >> 4;

  // A gather: thread loads row (tid&127), 16 floats at column chunk (tid>>7)*16
  const int arow = tid & 127;
  const int akc = (tid >> 7) * 16;
  size_t srctok = (MODE == 0) ? map_q_row(bm * 128 + arow) : map_kv_row(bm * 128 + arow);
  const float* aptr = A + srctok * 512 + akc;

  // B: thread loads 16 contiguous floats of W row (kk + tid>>3), cols (tid&7)*16..
  const int bkr = tid >> 3;
  const int bnc = (tid & 7) * 16;
  const float* bbase = W + (size_t)bn * 128 + bnc;

  v8f acc[2][4];
#pragma unroll
  for (int mt = 0; mt < 2; ++mt)
#pragma unroll
    for (int nt = 0; nt < 4; ++nt) acc[mt][nt] = (v8f){};

  for (int kk = 0; kk < 512; kk += 32) {
    const float4* ap = (const float4*)(aptr + kk);
#pragma unroll
    for (int j = 0; j < 4; ++j) {
      float4 f = ap[j];
      As[arow][akc + 4 * j + 0] = (_Float16)f.x;
      As[arow][akc + 4 * j + 1] = (_Float16)f.y;
      As[arow][akc + 4 * j + 2] = (_Float16)f.z;
      As[arow][akc + 4 * j + 3] = (_Float16)f.w;
    }
    const float4* bp = (const float4*)(bbase + (size_t)(kk + bkr) * N);
#pragma unroll
    for (int j = 0; j < 4; ++j) {
      float4 f = bp[j];
      Bs[bnc + 4 * j + 0][bkr] = (_Float16)f.x;
      Bs[bnc + 4 * j + 1][bkr] = (_Float16)f.y;
      Bs[bnc + 4 * j + 2][bkr] = (_Float16)f.z;
      Bs[bnc + 4 * j + 3][bkr] = (_Float16)f.w;
    }
    __syncthreads();

    v16h afr[2];
#pragma unroll
    for (int mt = 0; mt < 2; ++mt) {
      int m = wm * 32 + mt * 16 + lr;
      v8h lo = *(const v8h*)&As[m][hi * 8];
      v8h hh = *(const v8h*)&As[m][hi * 8 + 16];
      afr[mt] = combine16(lo, hh);
    }
#pragma unroll
    for (int nt = 0; nt < 4; ++nt) {
      int n = wn * 64 + nt * 16 + lr;
      v8h lo = *(const v8h*)&Bs[n][hi * 8];
      v8h hh = *(const v8h*)&Bs[n][hi * 8 + 16];
      v16h bfr = combine16(lo, hh);
#pragma unroll
      for (int mt = 0; mt < 2; ++mt) acc[mt][nt] = WMMA_F16(afr[mt], bfr, acc[mt][nt]);
    }
    __syncthreads();
  }

#pragma unroll
  for (int mt = 0; mt < 2; ++mt)
#pragma unroll
    for (int nt = 0; nt < 4; ++nt) {
      int gcol = bn * 128 + wn * 64 + nt * 16 + lr;
      float bz = bias[gcol];
#pragma unroll
      for (int i = 0; i < 8; ++i) {
        int gr = bm * 128 + wm * 32 + mt * 16 + i + 8 * hi;
        float v = acc[mt][nt][i] + bz;
        if (MODE == 0) {
          v *= 0.17677669529663687f;  // hd^-0.5
          out0[(size_t)gr * 512 + gcol] = (_Float16)v;
        } else {
          if (gcol < 512) out0[(size_t)gr * 512 + gcol] = (_Float16)v;
          else            out1[(size_t)gr * 512 + (gcol - 512)] = (_Float16)v;
        }
      }
    }
}

// ---------------------------------------------------------------------------
// Attention: one block per (window, head). 128 thr = 4 waves; wave = 16 q rows.
// logits 16x128 in wmma accumulators, + rpb + mask, shfl-based softmax,
// probs -> LDS f16, second wmma pass vs V^T.
// ---------------------------------------------------------------------------
__global__ __launch_bounds__(128) void attn_kernel(const _Float16* __restrict__ Qh,
                                                   const _Float16* __restrict__ Kh,
                                                   const _Float16* __restrict__ Vh,
                                                   const float* __restrict__ rpb_table,
                                                   _Float16* __restrict__ Oh) {
  __shared__ __align__(16) _Float16 Qs[64][40];   // [q][d]
  __shared__ __align__(16) _Float16 Ks[128][40];  // [key][d]
  __shared__ __align__(16) _Float16 Vs[32][136];  // [d][key]   (B-fragment layout)
  __shared__ __align__(16) _Float16 Af[64][136];  // [q][key]   softmax probs
  __shared__ float rpbs[232];

  const int wid = blockIdx.x >> 4;   // window
  const int h   = blockIdx.x & 15;   // head
  const int wh  = (wid >> 4) & 15;
  const int ww  = wid & 15;
  const int tid = threadIdx.x;
  const int lane = tid & 31;
  const int wv = tid >> 5;
  const int lr = lane & 15, hi = lane >> 4;

  // ---- stage Q / K / V / rpb column into LDS ----
  {
    int qrow = tid >> 1, qhalf = tid & 1;
    const v8h* qp = (const v8h*)(Qh + ((size_t)wid * 64 + qrow) * 512 + h * 32 + qhalf * 16);
    *(v8h*)&Qs[qrow][qhalf * 16] = qp[0];

    const v8h* kp = (const v8h*)(Kh + ((size_t)wid * 128 + tid) * 512 + h * 32);
    *(v8h*)&Ks[tid][0]  = kp[0];
    *(v8h*)&Ks[tid][8]  = kp[1];
    *(v8h*)&Ks[tid][16] = kp[2];
    *(v8h*)&Ks[tid][24] = kp[3];

    const v8h* vp = (const v8h*)(Vh + ((size_t)wid * 128 + tid) * 512 + h * 32);
    v8h v0 = vp[0], v1 = vp[1], v2 = vp[2], v3 = vp[3];
#pragma unroll
    for (int d = 0; d < 8; ++d) {
      Vs[d][tid]      = v0[d];
      Vs[8 + d][tid]  = v1[d];
      Vs[16 + d][tid] = v2[d];
      Vs[24 + d][tid] = v3[d];
    }
    for (int i = tid; i < 225; i += 128) rpbs[i] = rpb_table[i * 16 + h];
  }
  __syncthreads();

  // ---- logits = q @ k^T   (M=16 per wave, N=128, K=32 single wmma step) ----
  v8f acc[8];
#pragma unroll
  for (int nt = 0; nt < 8; ++nt) acc[nt] = (v8f){};
  {
    int m = wv * 16 + lr;
    v8h lo = *(const v8h*)&Qs[m][hi * 8];
    v8h hh = *(const v8h*)&Qs[m][hi * 8 + 16];
    v16h afr = combine16(lo, hh);
#pragma unroll
    for (int nt = 0; nt < 8; ++nt) {
      int n = nt * 16 + lr;
      v8h bl = *(const v8h*)&Ks[n][hi * 8];
      v8h bh = *(const v8h*)&Ks[n][hi * 8 + 16];
      acc[nt] = WMMA_F16(afr, combine16(bl, bh), acc[nt]);
    }
  }

  // ---- + relative position bias + shifted-window mask (in registers) ----
  const int qbase = wv * 16 + 8 * hi;  // accumulator element i holds row qbase+i
#pragma unroll
  for (int nt = 0; nt < 8; ++nt) {
    int n = nt * 16 + lr;
    int kqi = n & 63;                          // key index within window (tiled over nf)
    int rk = 3 * reg1(wh * 8 + (kqi >> 3)) + reg1(ww * 8 + (kqi & 7));
    int ck = rpc(63 - kqi);                    // flipped key coordinate
#pragma unroll
    for (int i = 0; i < 8; ++i) {
      int q = qbase + i;
      int rq = 3 * reg1(wh * 8 + (q >> 3)) + reg1(ww * 8 + (q & 7));
      float v = acc[nt][i] + rpbs[ck + rpc(q)] + ((rq != rk) ? -100.0f : 0.0f);
      acc[nt][i] = v;
    }
  }

  // ---- softmax over 128 keys: row spans 16 lanes (same hi) x 8 accums ----
#pragma unroll
  for (int i = 0; i < 8; ++i) {
    float m = acc[0][i];
#pragma unroll
    for (int nt = 1; nt < 8; ++nt) m = fmaxf(m, acc[nt][i]);
    m = fmaxf(m, __shfl_xor(m, 1));
    m = fmaxf(m, __shfl_xor(m, 2));
    m = fmaxf(m, __shfl_xor(m, 4));
    m = fmaxf(m, __shfl_xor(m, 8));
    float s = 0.0f;
#pragma unroll
    for (int nt = 0; nt < 8; ++nt) {
      float e = __expf(acc[nt][i] - m);
      acc[nt][i] = e;
      s += e;
    }
    s += __shfl_xor(s, 1);
    s += __shfl_xor(s, 2);
    s += __shfl_xor(s, 4);
    s += __shfl_xor(s, 8);
    float inv = 1.0f / s;
    int q = qbase + i;
#pragma unroll
    for (int nt = 0; nt < 8; ++nt) Af[q][nt * 16 + lr] = (_Float16)(acc[nt][i] * inv);
  }
  __syncthreads();

  // ---- out = probs @ V   (M=16 per wave, N=32, K=128 -> 4 wmma steps) ----
  v8f o[2];
  o[0] = (v8f){};
  o[1] = (v8f){};
#pragma unroll
  for (int kk = 0; kk < 128; kk += 32) {
    int m = wv * 16 + lr;
    v8h lo = *(const v8h*)&Af[m][kk + hi * 8];
    v8h hh = *(const v8h*)&Af[m][kk + hi * 8 + 16];
    v16h afr = combine16(lo, hh);
#pragma unroll
    for (int nt = 0; nt < 2; ++nt) {
      v8h bl = *(const v8h*)&Vs[nt * 16 + lr][kk + hi * 8];
      v8h bh = *(const v8h*)&Vs[nt * 16 + lr][kk + hi * 8 + 16];
      o[nt] = WMMA_F16(afr, combine16(bl, bh), o[nt]);
    }
  }
#pragma unroll
  for (int nt = 0; nt < 2; ++nt)
#pragma unroll
    for (int i = 0; i < 8; ++i) {
      int q = qbase + i;
      Oh[((size_t)wid * 64 + q) * 512 + h * 32 + nt * 16 + lr] = (_Float16)o[nt][i];
    }
}

// ---------------------------------------------------------------------------
// Output projection: f16 A (window order) x fp32 Wo -> fp32, scattered back
// through the inverse window/shift permutation. M=131072, N=512, K=512.
// ---------------------------------------------------------------------------
__global__ __launch_bounds__(256) void gemm_out(const _Float16* __restrict__ A,
                                                const float* __restrict__ W,
                                                const float* __restrict__ bias,
                                                float* __restrict__ out) {
  __shared__ __align__(16) _Float16 As[128][40];
  __shared__ __align__(16) _Float16 Bs[128][40];

  const int bm = blockIdx.x >> 2;   // 1024 m-tiles
  const int bn = blockIdx.x & 3;    // 4 n-tiles
  const int tid = threadIdx.x;
  const int lane = tid & 31;
  const int wv = tid >> 5;
  const int wm = wv >> 1, wn = wv & 1;
  const int lr = lane & 15, hi = lane >> 4;

  const int arow = tid & 127;
  const int akc = (tid >> 7) * 16;
  const _Float16* aptr = A + (size_t)(bm * 128 + arow) * 512 + akc;

  const int bkr = tid >> 3;
  const int bnc = (tid & 7) * 16;
  const float* bbase = W + (size_t)bn * 128 + bnc;

  v8f acc[2][4];
#pragma unroll
  for (int mt = 0; mt < 2; ++mt)
#pragma unroll
    for (int nt = 0; nt < 4; ++nt) acc[mt][nt] = (v8f){};

  for (int kk = 0; kk < 512; kk += 32) {
    const v8h* ap = (const v8h*)(aptr + kk);
    *(v8h*)&As[arow][akc]     = ap[0];
    *(v8h*)&As[arow][akc + 8] = ap[1];

    const float4* bp = (const float4*)(bbase + (size_t)(kk + bkr) * 512);
#pragma unroll
    for (int j = 0; j < 4; ++j) {
      float4 f = bp[j];
      Bs[bnc + 4 * j + 0][bkr] = (_Float16)f.x;
      Bs[bnc + 4 * j + 1][bkr] = (_Float16)f.y;
      Bs[bnc + 4 * j + 2][bkr] = (_Float16)f.z;
      Bs[bnc + 4 * j + 3][bkr] = (_Float16)f.w;
    }
    __syncthreads();

    v16h afr[2];
#pragma unroll
    for (int mt = 0; mt < 2; ++mt) {
      int m = wm * 32 + mt * 16 + lr;
      v8h lo = *(const v8h*)&As[m][hi * 8];
      v8h hh = *(const v8h*)&As[m][hi * 8 + 16];
      afr[mt] = combine16(lo, hh);
    }
#pragma unroll
    for (int nt = 0; nt < 4; ++nt) {
      int n = wn * 64 + nt * 16 + lr;
      v8h lo = *(const v8h*)&Bs[n][hi * 8];
      v8h hh = *(const v8h*)&Bs[n][hi * 8 + 16];
      v16h bfr = combine16(lo, hh);
#pragma unroll
      for (int mt = 0; mt < 2; ++mt) acc[mt][nt] = WMMA_F16(afr[mt], bfr, acc[mt][nt]);
    }
    __syncthreads();
  }

#pragma unroll
  for (int mt = 0; mt < 2; ++mt)
#pragma unroll
    for (int nt = 0; nt < 4; ++nt) {
      int gcol = bn * 128 + wn * 64 + nt * 16 + lr;
      float bz = bias[gcol];
#pragma unroll
      for (int i = 0; i < 8; ++i) {
        int gr = bm * 128 + wm * 32 + mt * 16 + i + 8 * hi;
        size_t tok = map_q_row(gr);  // same permutation forward == scatter back
        out[tok * 512 + gcol] = acc[mt][nt][i] + bz;
      }
    }
}

// ---------------------------------------------------------------------------
extern "C" void kernel_launch(void* const* d_in, const int* in_sizes, int n_in,
                              void* d_out, int out_size, void* d_ws, size_t ws_size,
                              hipStream_t stream) {
  (void)in_sizes; (void)n_in; (void)out_size; (void)ws_size;
  const float* q   = (const float*)d_in[0];
  const float* kv  = (const float*)d_in[1];
  const float* rpb = (const float*)d_in[4];
  const float* Wq  = (const float*)d_in[5];
  const float* bq  = (const float*)d_in[6];
  const float* Wkv = (const float*)d_in[7];
  const float* bkv = (const float*)d_in[8];
  const float* Wo  = (const float*)d_in[9];
  const float* bo  = (const float*)d_in[10];
  float* out = (float*)d_out;

  char* ws = (char*)d_ws;
  _Float16* Qh = (_Float16*)(ws);                                      // 131072*512 f16
  _Float16* Kh = (_Float16*)(ws + (size_t)134217728);                  // 262144*512 f16
  _Float16* Vh = (_Float16*)(ws + (size_t)134217728 + 268435456);      // 262144*512 f16
  _Float16* Oh = (_Float16*)(ws + (size_t)134217728 + 2 * (size_t)268435456);

  gemm_proj<0, 512><<<dim3(4096), dim3(256), 0, stream>>>(q, Wq, bq, Qh, nullptr);
  gemm_proj<1, 1024><<<dim3(16384), dim3(256), 0, stream>>>(kv, Wkv, bkv, Kh, Vh);
  attn_kernel<<<dim3(32768), dim3(128), 0, stream>>>(Qh, Kh, Vh, rpb, Oh);
  gemm_out<<<dim3(4096), dim3(256), 0, stream>>>(Oh, Wo, bo, out);
}
